// AttentionBlock_28724741275800
// MI455X (gfx1250) — compile-verified, matrix operand order checked by reading, not run on hardware
//
#include <hip/hip_runtime.h>

// ---- CDNA5 (gfx1250) wave32 WMMA attention block ------------------------
// x:[2,2048,1024] f32, w_qkv:[3072,1024] f32, w_out:[1024,1024] f32, b_out:[1024] f32
// out:[2,2048,1024] f32

#define DIMM   1024
#define NHEADS 16
#define DHEAD  64
#define INNERD 1024
#define BATCH  2
#define SEQ    2048

typedef __bf16 v16bf __attribute__((ext_vector_type(16)));
typedef float  v8f   __attribute__((ext_vector_type(8)));
typedef __bf16 bf8_t __attribute__((ext_vector_type(8)));
typedef int    v4i_t __attribute__((vector_size(4 * sizeof(int))));

#if defined(__gfx1250__) && __has_builtin(__builtin_amdgcn_global_load_async_to_lds_b128)
#define HAVE_ASYNC_LDS 1
#else
#define HAVE_ASYNC_LDS 0
#endif

union FragU { v16bf v; bf8_t h[2]; };

// Load a 16x32 bf16 A-fragment (or 32x16 B-fragment: lane owns row/col, K contiguous).
// ISA layout: lanes 0-15 -> K 0..7 (vgpr0-3), K 16..23 (vgpr4-7);
//             lanes 16-31 -> K 8..15, K 24..31.
__device__ inline v16bf load_frag(const __bf16* tile, int ld) {
  const int lane = threadIdx.x & 31;
  const int row  = lane & 15;
  const int half = lane >> 4;
  const __bf16* r = tile + row * ld + half * 8;
  FragU f;
  f.h[0] = *(const bf8_t*)(r);
  f.h[1] = *(const bf8_t*)(r + 16);
  return f.v;
}

__device__ inline v8f wmma_bf16(v16bf a, v16bf b, v8f c) {
  return __builtin_amdgcn_wmma_f32_16x16x32_bf16(
      false, a, false, b, (short)0, c, false, false);
}

// ---- fp32 -> bf16 convert ------------------------------------------------
__global__ void cvt_f32_bf16(const float* __restrict__ src, __bf16* __restrict__ dst, int n) {
  int i = blockIdx.x * blockDim.x + threadIdx.x;
  if (i < n) dst[i] = (__bf16)src[i];
}

// ---- QKV projection: [4096,1024] x [3072,1024]^T, scatter to q/k/v ------
// One wave computes 32(M) x 64(N); K loop unrolled x2 with ping-pong
// fragment sets (no register copies), one 32-deep K-step of lookahead.
__global__ void __launch_bounds__(32)
qkv_gemm(const __bf16* __restrict__ xb, const __bf16* __restrict__ wb,
         __bf16* __restrict__ qb, __bf16* __restrict__ kb, __bf16* __restrict__ vb) {
  const int mbase = blockIdx.x * 32;
  const int ntile = blockIdx.y * 64;
  v8f acc[2][4] = {};
  const __bf16* a0p = xb + (size_t)mbase * DIMM;
  const __bf16* a1p = xb + (size_t)(mbase + 16) * DIMM;
  const __bf16* bp  = wb + (size_t)ntile * DIMM;

  v16bf aA[2], bA[4], aB[2], bB[4];
  aA[0] = load_frag(a0p, DIMM);
  aA[1] = load_frag(a1p, DIMM);
#pragma unroll
  for (int ns = 0; ns < 4; ++ns) bA[ns] = load_frag(bp + (size_t)ns * 16 * DIMM, DIMM);

  for (int k0 = 0; k0 < DIMM; k0 += 64) {
    const int k1 = k0 + 32;
    aB[0] = load_frag(a0p + k1, DIMM);
    aB[1] = load_frag(a1p + k1, DIMM);
#pragma unroll
    for (int ns = 0; ns < 4; ++ns) bB[ns] = load_frag(bp + (size_t)ns * 16 * DIMM + k1, DIMM);
#pragma unroll
    for (int ns = 0; ns < 4; ++ns) {
      acc[0][ns] = wmma_bf16(aA[0], bA[ns], acc[0][ns]);
      acc[1][ns] = wmma_bf16(aA[1], bA[ns], acc[1][ns]);
    }
    const int k2 = (k0 + 64 < DIMM) ? (k0 + 64) : 0;   // harmless wrap at end
    aA[0] = load_frag(a0p + k2, DIMM);
    aA[1] = load_frag(a1p + k2, DIMM);
#pragma unroll
    for (int ns = 0; ns < 4; ++ns) bA[ns] = load_frag(bp + (size_t)ns * 16 * DIMM + k2, DIMM);
#pragma unroll
    for (int ns = 0; ns < 4; ++ns) {
      acc[0][ns] = wmma_bf16(aB[0], bB[ns], acc[0][ns]);
      acc[1][ns] = wmma_bf16(aB[1], bB[ns], acc[1][ns]);
    }
  }

  const int lane = threadIdx.x & 31;
  const int half = lane >> 4;
  const int col0 = lane & 15;
#pragma unroll
  for (int ns = 0; ns < 4; ++ns) {
    const int e     = ntile + ns * 16 + col0;   // 0..3071
    const int which = e >> 10;                  // 0=q 1=k 2=v
    const int inner = e & 1023;
    const int h  = inner >> 6;
    const int dh = inner & 63;
    __bf16* dst = (which == 0) ? qb : ((which == 1) ? kb : vb);
    const float scale = (which == 0) ? 0.125f : 1.0f;  // fold 1/sqrt(64) into q
#pragma unroll
    for (int mt = 0; mt < 2; ++mt) {
#pragma unroll
      for (int v = 0; v < 8; ++v) {
        const int m  = mbase + mt * 16 + v + 8 * half;
        const int b_ = m >> 11;
        const int n  = m & 2047;
        dst[((((size_t)b_ * NHEADS + h) * SEQ) + n) * DHEAD + dh] =
            (__bf16)(acc[mt][ns][v] * scale);
      }
    }
  }
}

// ---- Flash attention: per (b,h), 64 q-rows / block (4 waves), 64-key steps.
// K tile: double-buffered LDS, filled by gfx1250 async global->LDS DMA
// (ASYNCcnt); V tile: transposed at LDS store via register staging.
__global__ void __launch_bounds__(128)
flash_attn(const __bf16* __restrict__ qb, const __bf16* __restrict__ kb,
           const __bf16* __restrict__ vb, __bf16* __restrict__ ob) {
  __shared__ __bf16 k_tile[2][64 * DHEAD];  // [buf][key][dh]
  __shared__ __bf16 v_t[DHEAD * 64];        // [dh][key]
  __shared__ __bf16 p_tile[4][16 * 64];     // per-wave P (row-major, K=keys)

  const int bh    = blockIdx.y;             // b*16 + h
  const int wave  = threadIdx.x >> 5;
  const int lane  = threadIdx.x & 31;
  const int half  = lane >> 4;
  const int col0  = lane & 15;
  const int qbase = blockIdx.x * 64 + wave * 16;
  const int tid   = threadIdx.x;

  const __bf16* qrow = qb + ((size_t)bh * SEQ + qbase) * DHEAD;
  const v16bf aq0 = load_frag(qrow, DHEAD);        // dh 0..31
  const v16bf aq1 = load_frag(qrow + 32, DHEAD);   // dh 32..63

  v8f o[4] = {};
  float mrow[8], lrow[8];
#pragma unroll
  for (int v = 0; v < 8; ++v) { mrow[v] = -3.0e38f; lrow[v] = 0.0f; }

  const bf8_t* kg = (const bf8_t*)(kb + (size_t)bh * SEQ * DHEAD);
  const bf8_t* vg = (const bf8_t*)(vb + (size_t)bh * SEQ * DHEAD);

  bf8_t vreg[4];
#if !HAVE_ASYNC_LDS
  bf8_t kreg[4];
#endif

  // issue K chunk0 -> buf0, V chunk0 -> regs (64 keys = 512 bf8 chunks, 4/thread)
#if HAVE_ASYNC_LDS
#pragma unroll
  for (int e = 0; e < 4; ++e) {
    const int c = e * 128 + tid;
    __builtin_amdgcn_global_load_async_to_lds_b128(
        (v4i_t*)(kg + c), (v4i_t*)(&((bf8_t*)k_tile[0])[c]), 0, 0);
  }
#else
#pragma unroll
  for (int e = 0; e < 4; ++e) {
    const int c = e * 128 + tid;
    kreg[e] = kg[c];
  }
#pragma unroll
  for (int e = 0; e < 4; ++e) ((bf8_t*)k_tile[0])[e * 128 + tid] = kreg[e];
#endif
#pragma unroll
  for (int e = 0; e < 4; ++e) vreg[e] = vg[e * 128 + tid];

  for (int j = 0; j < SEQ; j += 64) {
    const int buf = (j >> 6) & 1;
    // (a) store V chunk j (transposed) into v_t
#pragma unroll
    for (int e = 0; e < 4; ++e) {
      const int c = e * 128 + tid;
      const int key = c >> 3, dhc = (c & 7) * 8;
#pragma unroll
      for (int i = 0; i < 8; ++i) v_t[(dhc + i) * 64 + key] = vreg[e][i];
    }
#if HAVE_ASYNC_LDS
    asm volatile("s_wait_asynccnt 0x0" ::: "memory");  // K chunk j landed in LDS
#endif
    __syncthreads();

    // (d) start next chunk: K -> other LDS buffer, V -> regs
    if (j + 64 < SEQ) {
      const int base = ((j + 64) * DHEAD) / 8;
#if HAVE_ASYNC_LDS
#pragma unroll
      for (int e = 0; e < 4; ++e) {
        const int c = e * 128 + tid;
        __builtin_amdgcn_global_load_async_to_lds_b128(
            (v4i_t*)(kg + base + c),
            (v4i_t*)(&((bf8_t*)k_tile[buf ^ 1])[c]), 0, 0);
      }
#else
#pragma unroll
      for (int e = 0; e < 4; ++e) kreg[e] = kg[base + e * 128 + tid];
#pragma unroll
      for (int e = 0; e < 4; ++e) ((bf8_t*)k_tile[buf ^ 1])[e * 128 + tid] = kreg[e];
#endif
#pragma unroll
      for (int e = 0; e < 4; ++e) vreg[e] = vg[base + e * 128 + tid];
    }

    // (e) S = Q K^T for 64 keys -> four 16x16 f32 tiles
    const __bf16* kt = k_tile[buf];
    v8f s[4];
#pragma unroll
    for (int t = 0; t < 4; ++t) {
      v8f sv = {};
      sv = wmma_bf16(aq0, load_frag(kt + t * 16 * DHEAD, DHEAD), sv);
      sv = wmma_bf16(aq1, load_frag(kt + t * 16 * DHEAD + 32, DHEAD), sv);
      s[t] = sv;
    }

    // online softmax across these 64 keys
    __bf16* pw = &p_tile[wave][0];
#pragma unroll
    for (int v = 0; v < 8; ++v) {
      float loc = fmaxf(fmaxf(s[0][v], s[1][v]), fmaxf(s[2][v], s[3][v]));
#pragma unroll
      for (int off = 1; off < 16; off <<= 1)     // stays inside 16-lane row group
        loc = fmaxf(loc, __shfl_xor(loc, off, 32));
      const float mn    = fmaxf(mrow[v], loc);
      const float alpha = __expf(mrow[v] - mn);
      const float p0 = __expf(s[0][v] - mn);
      const float p1 = __expf(s[1][v] - mn);
      const float p2 = __expf(s[2][v] - mn);
      const float p3 = __expf(s[3][v] - mn);
      float rs = (p0 + p1) + (p2 + p3);
#pragma unroll
      for (int off = 1; off < 16; off <<= 1)
        rs += __shfl_xor(rs, off, 32);
      lrow[v] = lrow[v] * alpha + rs;
      mrow[v] = mn;
      o[0][v] *= alpha; o[1][v] *= alpha; o[2][v] *= alpha; o[3][v] *= alpha;
      const int r = v + 8 * half;                // C-layout -> A-layout via LDS
      pw[r * 64 +  0 + col0] = (__bf16)p0;
      pw[r * 64 + 16 + col0] = (__bf16)p1;
      pw[r * 64 + 32 + col0] = (__bf16)p2;
      pw[r * 64 + 48 + col0] = (__bf16)p3;
    }

    // O += P(16x64) * V(64x64); same-wave LDS RAW kept in order by DScnt
    const v16bf ap0 = load_frag(pw, 64);         // keys 0..31
    const v16bf ap1 = load_frag(pw + 32, 64);    // keys 32..63
#pragma unroll
    for (int n0 = 0; n0 < 4; ++n0) {
      o[n0] = wmma_bf16(ap0, load_frag(v_t + n0 * 16 * 64, 64), o[n0]);
      o[n0] = wmma_bf16(ap1, load_frag(v_t + n0 * 16 * 64 + 32, 64), o[n0]);
    }
    __syncthreads();
  }

  // normalize and write [b, n, h, dh] so the out-proj sees row-major [m, inner]
  const int b_ = bh >> 4, h = bh & 15;
#pragma unroll
  for (int v = 0; v < 8; ++v) {
    const float inv = 1.0f / lrow[v];
    const int n = qbase + v + 8 * half;
    const size_t base = (((size_t)b_ * SEQ + n) * NHEADS + h) * DHEAD;
    ob[base +  0 + col0] = (__bf16)(o[0][v] * inv);
    ob[base + 16 + col0] = (__bf16)(o[1][v] * inv);
    ob[base + 32 + col0] = (__bf16)(o[2][v] * inv);
    ob[base + 48 + col0] = (__bf16)(o[3][v] * inv);
  }
}

// ---- Output projection: [4096,1024] x [1024,1024]^T + bias -> f32 -------
__global__ void __launch_bounds__(32)
out_gemm(const __bf16* __restrict__ ab, const __bf16* __restrict__ wb,
         const float* __restrict__ bias, float* __restrict__ out) {
  const int mbase = blockIdx.x * 32;
  const int ntile = blockIdx.y * 64;
  v8f acc[2][4] = {};
  const __bf16* a0p = ab + (size_t)mbase * INNERD;
  const __bf16* a1p = ab + (size_t)(mbase + 16) * INNERD;
  const __bf16* bp  = wb + (size_t)ntile * INNERD;

  v16bf aA[2], bA[4], aB[2], bB[4];
  aA[0] = load_frag(a0p, INNERD);
  aA[1] = load_frag(a1p, INNERD);
#pragma unroll
  for (int ns = 0; ns < 4; ++ns) bA[ns] = load_frag(bp + (size_t)ns * 16 * INNERD, INNERD);

  for (int k0 = 0; k0 < INNERD; k0 += 64) {
    const int k1 = k0 + 32;
    aB[0] = load_frag(a0p + k1, INNERD);
    aB[1] = load_frag(a1p + k1, INNERD);
#pragma unroll
    for (int ns = 0; ns < 4; ++ns) bB[ns] = load_frag(bp + (size_t)ns * 16 * INNERD + k1, INNERD);
#pragma unroll
    for (int ns = 0; ns < 4; ++ns) {
      acc[0][ns] = wmma_bf16(aA[0], bA[ns], acc[0][ns]);
      acc[1][ns] = wmma_bf16(aA[1], bA[ns], acc[1][ns]);
    }
    const int k2 = (k0 + 64 < INNERD) ? (k0 + 64) : 0;
    aA[0] = load_frag(a0p + k2, INNERD);
    aA[1] = load_frag(a1p + k2, INNERD);
#pragma unroll
    for (int ns = 0; ns < 4; ++ns) bA[ns] = load_frag(bp + (size_t)ns * 16 * INNERD + k2, INNERD);
#pragma unroll
    for (int ns = 0; ns < 4; ++ns) {
      acc[0][ns] = wmma_bf16(aB[0], bB[ns], acc[0][ns]);
      acc[1][ns] = wmma_bf16(aB[1], bB[ns], acc[1][ns]);
    }
  }

  const int lane = threadIdx.x & 31;
  const int half = lane >> 4;
  const int col0 = lane & 15;
#pragma unroll
  for (int ns = 0; ns < 4; ++ns) {
    const int c = ntile + ns * 16 + col0;
    const float bv = bias[c];
#pragma unroll
    for (int mt = 0; mt < 2; ++mt) {
#pragma unroll
      for (int v = 0; v < 8; ++v) {
        const int m = mbase + mt * 16 + v + 8 * half;
        out[(size_t)m * DIMM + c] = acc[mt][ns][v] + bv;
      }
    }
  }
}

// ---- host launch ---------------------------------------------------------
extern "C" void kernel_launch(void* const* d_in, const int* in_sizes, int n_in,
                              void* d_out, int out_size, void* d_ws, size_t ws_size,
                              hipStream_t stream) {
  (void)in_sizes; (void)n_in; (void)out_size; (void)ws_size;
  const float* x     = (const float*)d_in[0];
  const float* w_qkv = (const float*)d_in[1];
  const float* w_out = (const float*)d_in[2];
  const float* b_out = (const float*)d_in[3];

  char* ws = (char*)d_ws;
  size_t off = 0;
  auto alloc = [&](size_t elems) {
    __bf16* p = (__bf16*)(ws + off);
    off += elems * sizeof(__bf16);
    return p;
  };
  __bf16* xb    = alloc((size_t)BATCH * SEQ * DIMM);      // 8 MB
  __bf16* wqkvb = alloc((size_t)3 * INNERD * DIMM);       // 6 MB
  __bf16* woutb = alloc((size_t)DIMM * INNERD);           // 2 MB
  __bf16* qb    = alloc((size_t)BATCH * NHEADS * SEQ * DHEAD);
  __bf16* kb    = alloc((size_t)BATCH * NHEADS * SEQ * DHEAD);
  __bf16* vb    = alloc((size_t)BATCH * NHEADS * SEQ * DHEAD);
  __bf16* attb  = alloc((size_t)BATCH * SEQ * INNERD);    // total ~48 MB

  const int CT = 256;
  const int nx = BATCH * SEQ * DIMM;
  const int nw = 3 * INNERD * DIMM;
  const int no = DIMM * INNERD;
  cvt_f32_bf16<<<(nx + CT - 1) / CT, CT, 0, stream>>>(x, xb, nx);
  cvt_f32_bf16<<<(nw + CT - 1) / CT, CT, 0, stream>>>(w_qkv, wqkvb, nw);
  cvt_f32_bf16<<<(no + CT - 1) / CT, CT, 0, stream>>>(w_out, woutb, no);

  qkv_gemm<<<dim3((BATCH * SEQ) / 32, (3 * INNERD) / 64), 32, 0, stream>>>(
      xb, wqkvb, qb, kb, vb);

  flash_attn<<<dim3(SEQ / 64, BATCH * NHEADS), 128, 0, stream>>>(qb, kb, vb, attb);

  out_gemm<<<dim3((BATCH * SEQ) / 32, DIMM / 64), 32, 0, stream>>>(
      attb, woutb, b_out, (float*)d_out);
}
